// Simplicial_01_dense_Layer_90065464197252
// MI455X (gfx1250) — matrix operand order repaired and checked by reading, not compiled, not run on hardware
//
#include <hip/hip_runtime.h>
#include <hip/hip_bf16.h>

// Problem constants
#define B_ 4
#define S_ 1024
#define D_ 1024
#define H_ 16
#define HD_ 64
#define M_TOK 4096          // B_*S_
#define SCALE_ 0.125f       // HD^-0.5 = 64^-0.5

typedef __attribute__((ext_vector_type(16))) __bf16      v16bf;
typedef __attribute__((ext_vector_type(8)))  float       v8f;
typedef __attribute__((ext_vector_type(4)))  unsigned int v4u;

union BF16Frag {
    v16bf v;
    v4u   q[2];
    unsigned short u[16];
};

__device__ __forceinline__ unsigned short f2bf(float f) {
    unsigned int x = __float_as_uint(f);
    unsigned int r = 0x7FFFu + ((x >> 16) & 1u);   // round-to-nearest-even
    return (unsigned short)((x + r) >> 16);
}

// ---------------------------------------------------------------------------
// fp32 -> bf16 conversion (grid-stride). Inputs are read once -> NT loads.
// ---------------------------------------------------------------------------
__global__ void convert_f32_to_bf16(const float* __restrict__ src,
                                    unsigned short* __restrict__ dst, int n) {
    int i = blockIdx.x * blockDim.x + threadIdx.x;
    int stride = gridDim.x * blockDim.x;
    for (; i < n; i += stride) dst[i] = f2bf(__builtin_nontemporal_load(&src[i]));
}

// ---------------------------------------------------------------------------
// C[M,N] = A[M,K]bf16 @ W[N,K]^T bf16 + bias[N], then scale; layout per mode:
//   mode 0/1: out bf16 [B,H,S,HD]   (Q with scale, K)
//   mode 2  : out bf16 [B,H,HD,S]   (V transposed for P*V B-operand)
//   mode 3  : out fp32 [M,N]        (final projection -> d_out, NT stores)
// Block: 256 threads = 8 waves; tile 64(M) x 64(N); wave -> 16x32.
// ---------------------------------------------------------------------------
__global__ __launch_bounds__(256)
void gemm_qkvo(const unsigned short* __restrict__ A,
               const unsigned short* __restrict__ W,
               const float* __restrict__ bias,
               void* __restrict__ out,
               float scale, int mode) {
    const int lane = threadIdx.x & 31;
    const int wave = threadIdx.x >> 5;
    const int n    = lane & 15;   // col within 16x16 tile; also A-frag row id
    const int half = lane >> 4;

    const int mbase = blockIdx.y * 64 + (wave & 3) * 16;
    const int nbase = blockIdx.x * 64 + (wave >> 2) * 32;

    v8f c0 = {}, c1 = {};

    const unsigned short* arow  = A + (size_t)(mbase + n)       * D_;
    const unsigned short* wrow0 = W + (size_t)(nbase + n)       * D_;
    const unsigned short* wrow1 = W + (size_t)(nbase + 16 + n)  * D_;

    for (int kk = 0; kk < D_; kk += 32) {
        BF16Frag a, b0, b1;
        // A 16x32 bf16 fragment: lane half 0 -> K {0..15}, half 1 -> {16..31}
        a.q[0]  = *(const v4u*)(arow  + kk + half * 16);
        a.q[1]  = *(const v4u*)(arow  + kk + half * 16 + 8);
        // B 32x16 fragment: lane = N col, K = half*16 .. half*16+15
        b0.q[0] = *(const v4u*)(wrow0 + kk + half * 16);
        b0.q[1] = *(const v4u*)(wrow0 + kk + half * 16 + 8);
        b1.q[0] = *(const v4u*)(wrow1 + kk + half * 16);
        b1.q[1] = *(const v4u*)(wrow1 + kk + half * 16 + 8);
        c0 = __builtin_amdgcn_wmma_f32_16x16x32_bf16(false, a.v, false, b0.v,
                                                     (short)0, c0, false, false);
        c1 = __builtin_amdgcn_wmma_f32_16x16x32_bf16(false, a.v, false, b1.v,
                                                     (short)0, c1, false, false);
    }

    const float bn0 = bias[nbase + n];
    const float bn1 = bias[nbase + 16 + n];

    for (int r = 0; r < 8; ++r) {
        const int m  = mbase + r + 8 * half;    // global token
        const int bb = m >> 10;                 // batch
        const int s  = m & 1023;                // sequence pos
        const float v0 = (c0[r] + bn0) * scale;
        const float v1 = (c1[r] + bn1) * scale;
        const int nc0 = nbase + n;
        const int nc1 = nbase + 16 + n;
        if (mode == 3) {
            float* o = (float*)out;             // final output: written once
            __builtin_nontemporal_store(v0, &o[(size_t)m * D_ + nc0]);
            __builtin_nontemporal_store(v1, &o[(size_t)m * D_ + nc1]);
        } else if (mode == 2) {
            unsigned short* o = (unsigned short*)out;   // Vt [B,H,HD,S]
            o[(size_t)((bb * H_ + (nc0 >> 6)) * HD_ + (nc0 & 63)) * S_ + s] = f2bf(v0);
            o[(size_t)((bb * H_ + (nc1 >> 6)) * HD_ + (nc1 & 63)) * S_ + s] = f2bf(v1);
        } else {
            unsigned short* o = (unsigned short*)out;   // [B,H,S,HD]
            o[(size_t)((bb * H_ + (nc0 >> 6)) * S_ + s) * HD_ + (nc0 & 63)] = f2bf(v0);
            o[(size_t)((bb * H_ + (nc1 >> 6)) * S_ + s) * HD_ + (nc1 & 63)] = f2bf(v1);
        }
    }
}

// ---------------------------------------------------------------------------
// Flash attention: softmax(Q K^T + bias) V with online softmax.
// Grid (S/64, H, B), block 128 = 4 waves; each wave owns 16 q rows.
// Score/accumulator kept in WMMA C-layout: lane(half) holds rows {8*half..+7},
// one element per VGPR, col = lane&15 -> row stats reduce across 16-lane half.
// The 256MB bias stream is touched once: non-temporal loads so it does not
// evict the K/V/Q working set (fits in the 192MB L2) while streaming.
// ---------------------------------------------------------------------------
__global__ __launch_bounds__(128)
void flash_attn(const unsigned short* __restrict__ Qh,
                const unsigned short* __restrict__ Kh,
                const unsigned short* __restrict__ Vt,
                const float* __restrict__ bias,
                unsigned short* __restrict__ AO) {
    __shared__ unsigned short sP[4][16][32];    // per-wave P tile (16q x 32k)

    const int lane = threadIdx.x & 31;
    const int wave = threadIdx.x >> 5;
    const int n    = lane & 15;
    const int half = lane >> 4;
    const int h    = blockIdx.y;
    const int b    = blockIdx.z;
    const int bh   = b * H_ + h;
    const int qbase = blockIdx.x * 64 + wave * 16;

    // Q A-fragments for hd 0..31 and 32..63 (kept resident)
    const unsigned short* qrow = Qh + (size_t)(bh * S_ + qbase + n) * HD_;
    BF16Frag qa0, qa1;
    qa0.q[0] = *(const v4u*)(qrow + half * 16);
    qa0.q[1] = *(const v4u*)(qrow + half * 16 + 8);
    qa1.q[0] = *(const v4u*)(qrow + 32 + half * 16);
    qa1.q[1] = *(const v4u*)(qrow + 32 + half * 16 + 8);

    float m_i[8], l_i[8];
    v8f acc[4] = {};
    for (int r = 0; r < 8; ++r) { m_i[r] = -3.0e30f; l_i[r] = 0.0f; }

    const float* brow = bias + (size_t)bh * S_ * S_;

    for (int kb = 0; kb < S_; kb += 32) {
        // ---- scores: two 16-key sub-tiles, K-dim = HD=64 -> 2 WMMAs each
        v8f s0 = {}, s1 = {};
        {
            const unsigned short* kr0 = Kh + (size_t)(bh * S_ + kb + n) * HD_;
            const unsigned short* kr1 = kr0 + 16 * HD_;
            BF16Frag klo, khi;
            klo.q[0] = *(const v4u*)(kr0 + half * 16);
            klo.q[1] = *(const v4u*)(kr0 + half * 16 + 8);
            khi.q[0] = *(const v4u*)(kr0 + 32 + half * 16);
            khi.q[1] = *(const v4u*)(kr0 + 32 + half * 16 + 8);
            s0 = __builtin_amdgcn_wmma_f32_16x16x32_bf16(false, qa0.v, false, klo.v,
                                                         (short)0, s0, false, false);
            s0 = __builtin_amdgcn_wmma_f32_16x16x32_bf16(false, qa1.v, false, khi.v,
                                                         (short)0, s0, false, false);
            klo.q[0] = *(const v4u*)(kr1 + half * 16);
            klo.q[1] = *(const v4u*)(kr1 + half * 16 + 8);
            khi.q[0] = *(const v4u*)(kr1 + 32 + half * 16);
            khi.q[1] = *(const v4u*)(kr1 + 32 + half * 16 + 8);
            s1 = __builtin_amdgcn_wmma_f32_16x16x32_bf16(false, qa0.v, false, klo.v,
                                                         (short)0, s1, false, false);
            s1 = __builtin_amdgcn_wmma_f32_16x16x32_bf16(false, qa1.v, false, khi.v,
                                                         (short)0, s1, false, false);
        }

        // ---- + bias (non-temporal stream), online softmax over the 32 keys
        float p0[8], p1[8], mx[8];
        for (int r = 0; r < 8; ++r) {
            const int q = qbase + r + 8 * half;
            p0[r] = s0[r] + __builtin_nontemporal_load(&brow[(size_t)q * S_ + kb + n]);
            p1[r] = s1[r] + __builtin_nontemporal_load(&brow[(size_t)q * S_ + kb + 16 + n]);
            mx[r] = fmaxf(p0[r], p1[r]);
        }
        for (int off = 1; off < 16; off <<= 1)
            for (int r = 0; r < 8; ++r)
                mx[r] = fmaxf(mx[r], __shfl_xor(mx[r], off, 32));

        float rs[8];
        for (int r = 0; r < 8; ++r) {
            const float nm   = fmaxf(m_i[r], mx[r]);
            const float corr = __expf(m_i[r] - nm);
            p0[r] = __expf(p0[r] - nm);
            p1[r] = __expf(p1[r] - nm);
            rs[r] = p0[r] + p1[r];
            m_i[r] = nm;
            l_i[r] *= corr;
            for (int dt = 0; dt < 4; ++dt) acc[dt][r] *= corr;
            sP[wave][r + 8 * half][n]      = f2bf(p0[r]);
            sP[wave][r + 8 * half][16 + n] = f2bf(p1[r]);
        }
        for (int off = 1; off < 16; off <<= 1)
            for (int r = 0; r < 8; ++r)
                rs[r] += __shfl_xor(rs[r], off, 32);
        for (int r = 0; r < 8; ++r) l_i[r] += rs[r];

        // same-wave LDS RAW: make P visible before re-reading in A layout
        asm volatile("s_wait_dscnt 0" ::: "memory");

        BF16Frag pa;
        pa.q[0] = *(const v4u*)(&sP[wave][n][half * 16]);
        pa.q[1] = *(const v4u*)(&sP[wave][n][half * 16 + 8]);

        // ---- accumulate P(16x32) @ V(32x16) for 4 hd tiles
        for (int dt = 0; dt < 4; ++dt) {
            const unsigned short* vr =
                Vt + (size_t)(bh * HD_ + dt * 16 + n) * S_ + kb + half * 16;
            BF16Frag vb;
            vb.q[0] = *(const v4u*)(vr);
            vb.q[1] = *(const v4u*)(vr + 8);
            acc[dt] = __builtin_amdgcn_wmma_f32_16x16x32_bf16(false, pa.v, false, vb.v,
                                                              (short)0, acc[dt], false, false);
        }
    }

    // ---- epilogue: normalize and store bf16 into attn_out [B,S,D]
    for (int r = 0; r < 8; ++r) {
        const int q = qbase + r + 8 * half;
        const float inv = 1.0f / l_i[r];
        unsigned short* orow = AO + (size_t)(b * S_ + q) * D_ + h * HD_;
        for (int dt = 0; dt < 4; ++dt)
            orow[dt * 16 + n] = f2bf(acc[dt][r] * inv);
    }
}

// ---------------------------------------------------------------------------
extern "C" void kernel_launch(void* const* d_in, const int* in_sizes, int n_in,
                              void* d_out, int out_size, void* d_ws, size_t ws_size,
                              hipStream_t stream) {
    const float* query = (const float*)d_in[0];
    const float* abias = (const float*)d_in[1];
    const float* Wq = (const float*)d_in[2];
    const float* bq = (const float*)d_in[3];
    const float* Wk = (const float*)d_in[4];
    const float* bk = (const float*)d_in[5];
    const float* Wv = (const float*)d_in[6];
    const float* bv = (const float*)d_in[7];
    const float* Wo = (const float*)d_in[8];
    const float* bo = (const float*)d_in[9];
    float* out = (float*)d_out;

    const size_t NTOK = (size_t)M_TOK * D_;   // 4M elements
    const size_t NW   = (size_t)D_ * D_;      // 1M elements

    unsigned short* ws  = (unsigned short*)d_ws;
    unsigned short* Xb  = ws;                 // query bf16      [4096,1024]
    unsigned short* Wqb = Xb  + NTOK;
    unsigned short* Wkb = Wqb + NW;
    unsigned short* Wvb = Wkb + NW;
    unsigned short* Wob = Wvb + NW;
    unsigned short* Qh  = Wob + NW;           // [B,H,S,HD] scaled
    unsigned short* Kh  = Qh  + NTOK;         // [B,H,S,HD]
    unsigned short* Vt  = Kh  + NTOK;         // [B,H,HD,S]
    unsigned short* AOb = Vt  + NTOK;         // attn out bf16 [4096,1024]

    // Stage 0: fp32 -> bf16
    convert_f32_to_bf16<<<4096, 256, 0, stream>>>(query, Xb, (int)NTOK);
    convert_f32_to_bf16<<<1024, 256, 0, stream>>>(Wq, Wqb, (int)NW);
    convert_f32_to_bf16<<<1024, 256, 0, stream>>>(Wk, Wkb, (int)NW);
    convert_f32_to_bf16<<<1024, 256, 0, stream>>>(Wv, Wvb, (int)NW);
    convert_f32_to_bf16<<<1024, 256, 0, stream>>>(Wo, Wob, (int)NW);

    // Stage 1: QKV projections (WMMA GEMMs)
    dim3 ggrid(D_ / 64, M_TOK / 64);          // (16, 64)
    gemm_qkvo<<<ggrid, 256, 0, stream>>>(Xb, Wqb, bq, Qh, SCALE_, 0);
    gemm_qkvo<<<ggrid, 256, 0, stream>>>(Xb, Wkb, bk, Kh, 1.0f, 1);
    gemm_qkvo<<<ggrid, 256, 0, stream>>>(Xb, Wvb, bv, Vt, 1.0f, 2);

    // Stage 2: flash attention (streams the 256MB bias once, NT)
    dim3 fgrid(S_ / 64, H_, B_);              // (16, 16, 4)
    flash_attn<<<fgrid, 128, 0, stream>>>(Qh, Kh, Vt, abias, AOb);

    // Stage 3: output projection -> fp32 d_out (NT stores)
    gemm_qkvo<<<ggrid, 256, 0, stream>>>(AOb, Wob, bo, out, 1.0f, 3);
}
